// Model_33062658245168
// MI455X (gfx1250) — matrix operand order repaired and checked by reading, not compiled, not run on hardware
//
#include <hip/hip_runtime.h>
#include <hip/hip_bf16.h>
#include <math.h>

typedef __attribute__((ext_vector_type(16))) _Float16 v16h;
typedef __attribute__((ext_vector_type(8)))  float    v8f;
typedef __attribute__((ext_vector_type(4)))  int      v4i;
typedef _Float16 h16;

#define ACT_NONE 0
#define ACT_RELU 1

constexpr int Bb = 32, Ss = 512, Cc = 512, Dd = 512, DFf = 2048, Hh = 8, Ee = 64;
constexpr int PREDp = 96, KTOP = 10;
constexpr int NR = Bb * Cc;   // 16384 rows for all big GEMMs

// ---------------- CDNA5 async / tensor data-mover availability probes ----------------
#if defined(__has_builtin)
# if __has_builtin(__builtin_amdgcn_global_load_async_to_lds_b128)
#  define HAS_ASYNC_LDS 1
# endif
# if __has_builtin(__builtin_amdgcn_tensor_load_to_lds)
#  define HAS_TDM 1
# endif
# if __has_builtin(__builtin_amdgcn_s_wait_asynccnt)
#  define WAIT_ASYNC0() __builtin_amdgcn_s_wait_asynccnt(0)
# endif
# if __has_builtin(__builtin_amdgcn_s_wait_tensorcnt)
#  define WAIT_TENSOR0() __builtin_amdgcn_s_wait_tensorcnt(0)
# endif
#endif
#ifndef WAIT_ASYNC0
# define WAIT_ASYNC0() asm volatile("s_wait_asynccnt 0x0" ::: "memory")
#endif
#ifndef WAIT_TENSOR0
# define WAIT_TENSOR0() asm volatile("s_wait_tensorcnt 0x0" ::: "memory")
#endif
#ifndef HAS_ASYNC_LDS
#warning "CDNA5 probe: __builtin_amdgcn_global_load_async_to_lds_b128 NOT available; GEMM uses sync LDS copy"
#endif
#ifndef HAS_TDM
#warning "CDNA5 probe: __builtin_amdgcn_tensor_load_to_lds NOT available; attention uses sync LDS copy"
#endif

// ---------------- wave reductions (wave32) ----------------
__device__ __forceinline__ float wave_rmax(float v) {
#pragma unroll
  for (int o = 16; o > 0; o >>= 1) v = fmaxf(v, __shfl_xor(v, o, 32));
  return v;
}
__device__ __forceinline__ float wave_rsum(float v) {
#pragma unroll
  for (int o = 16; o > 0; o >>= 1) v += __shfl_xor(v, o, 32);
  return v;
}

// ---------------- WMMA fragment helpers (contiguous k = 16*(lane/16) + j in A and B) ----------------
__device__ __forceinline__ v16h frag_a16x32(const h16* tile, int ld, int rowbase, int lane) {
  const h16* p = tile + (rowbase + (lane & 15)) * ld + ((lane >> 4) << 4);
  v16h a;
#pragma unroll
  for (int j = 0; j < 16; ++j) a[j] = p[j];
  return a;
}
__device__ __forceinline__ v16h frag_b32x16(const h16* tile, int ld, int colbase, int lane) {
  const h16* p = tile + (colbase + (lane & 15)) * ld + ((lane >> 4) << 4);
  v16h b;
#pragma unroll
  for (int j = 0; j < 16; ++j) b[j] = p[j];
  return b;
}
__device__ __forceinline__ v16h frag_b32x16_kn(const h16* tile, int ld, int colbase, int lane) {
  const h16* p = tile + (((lane >> 4) << 4)) * ld + (colbase + (lane & 15));
  v16h b;
#pragma unroll
  for (int j = 0; j < 16; ++j) b[j] = p[j * ld];
  return b;
}
__device__ __forceinline__ v8f wmma_f16(v16h a, v16h b, v8f c) {
  return __builtin_amdgcn_wmma_f32_16x16x32_f16(false, a, false, b, (short)0, c, false, false);
}

// ---------------- TDM helper: 2D tile (512 rows x 64 f16 cols, row stride 512) ----------------
#ifdef HAS_TDM
typedef __attribute__((ext_vector_type(4))) unsigned int u32x4;
typedef __attribute__((ext_vector_type(8))) int i32x8;
typedef __attribute__((ext_vector_type(4))) int i32x4;

__device__ __forceinline__ void tdm_load_tile_512x64(const h16* gsrc, h16* ldst) {
  unsigned long long ga = (unsigned long long)(uintptr_t)gsrc;
  unsigned lds_a = (unsigned)(uintptr_t)ldst;        // low 32 bits = LDS byte address
  u32x4 g0;
  g0[0] = 1u;                                        // count=1, user descriptor
  g0[1] = lds_a;                                     // lds_addr
  g0[2] = (unsigned)ga;                              // global_addr[31:0]
  g0[3] = (unsigned)((ga >> 32) & 0x01FFFFFFu) | (2u << 30);  // addr[56:32] | type=2
  i32x8 g1;
  g1[0] = 0x00010000;                                // data_size=1 (2 bytes/elem)
  g1[1] = (int)(512u << 16);                         // tensor_dim0=512 (bits 79:48 lo)
  g1[2] = (int)(16384u << 16);                       // dim0 hi=0 | tensor_dim1=16384 lo
  g1[3] = (int)(64u << 16);                          // dim1 hi=0 | tile_dim0=64
  g1[4] = 512;                                       // tile_dim1=512, tile_dim2=0
  g1[5] = 512;                                       // tensor_dim0_stride=512 (lo 32)
  g1[6] = 0;                                         // stride0 hi | dim1_stride lo
  g1[7] = 0;
  i32x4 gz4 = {0, 0, 0, 0};
  i32x8 gz8 = {0, 0, 0, 0, 0, 0, 0, 0};
  __builtin_amdgcn_tensor_load_to_lds(g0, g1, gz4, gz4, gz8, 0);
}
#endif

// ---------------- generic WMMA GEMM: out = ACT(A@W + bias (+res)) ----------------
// double-buffered LDS tiles; A-tile via async global->LDS (ASYNCcnt) when available
template<int ACT>
__global__ __launch_bounds__(256)
void gemm_wmma_kernel(const h16* __restrict__ A, const h16* __restrict__ W,
                      const float* __restrict__ bias, const float* __restrict__ res,
                      float* __restrict__ outf, h16* __restrict__ outh,
                      int M, int N, int K)
{
  constexpr int LDA = 32, LDB = 40;
  __shared__ h16 As[2][128 * LDA];          // [row][k] (16B-aligned rows for async)
  __shared__ h16 Bs[2][128 * LDB];          // [col][k] (transposed W tile)
  const int tid = threadIdx.x;
  const int wid = tid >> 5, lane = tid & 31;
  const int wm = wid >> 2, wn = wid & 3;    // 2 x 4 wave grid
  const int bm = blockIdx.y * 128, bn = blockIdx.x * 128;

  auto load_tile = [&](int k0, h16* Asb, h16* Bsb) {
    // A tile 128x32 : 512 chunks of 16B (rows always in range, M % 128 == 0)
#ifdef HAS_ASYNC_LDS
    for (int c = tid; c < 512; c += 256) {
      int row = c >> 2, ko = (c & 3) << 3;
      __builtin_amdgcn_global_load_async_to_lds_b128(
          (__attribute__((address_space(1))) v4i*)(uintptr_t)(A + (size_t)(bm + row) * K + k0 + ko),
          (__attribute__((address_space(3))) v4i*)(unsigned)(uintptr_t)(&Asb[row * LDA + ko]),
          0, 0);
    }
#else
    for (int c = tid; c < 512; c += 256) {
      int row = c >> 2, ko = (c & 3) << 3;
      float4 t = *(const float4*)(A + (size_t)(bm + row) * K + k0 + ko);
      *(float4*)(&Asb[row * LDA + ko]) = t;
    }
#endif
    // W tile 32x128 -> Bs[col][k] (transpose scatter), guard N
    for (int c = tid; c < 512; c += 256) {
      int kk = c >> 4, nc = (c & 15) << 3;
      int gn = bn + nc;
      h16 tv[8];
      if (gn + 7 < N) {
        float4 t = *(const float4*)(W + (size_t)(k0 + kk) * N + gn);
        const h16* th = (const h16*)&t;
#pragma unroll
        for (int j = 0; j < 8; ++j) tv[j] = th[j];
      } else {
#pragma unroll
        for (int j = 0; j < 8; ++j)
          tv[j] = (gn + j < N) ? W[(size_t)(k0 + kk) * N + gn + j] : (h16)0.f;
      }
#pragma unroll
      for (int j = 0; j < 8; ++j) Bsb[(nc + j) * LDB + kk] = tv[j];
    }
    // L2 prefetch of the tile after next
    if (k0 + 64 < K)
      __builtin_prefetch(A + (size_t)(bm + (tid >> 1)) * K + k0 + 64, 0, 2);
  };

  v8f zero = {};
  v8f acc[4][2];
#pragma unroll
  for (int mt = 0; mt < 4; ++mt) { acc[mt][0] = zero; acc[mt][1] = zero; }

  load_tile(0, As[0], Bs[0]);
  const int steps = K >> 5;
  for (int i = 0; i < steps; ++i) {
    WAIT_ASYNC0();            // this wave's async tile-i writes landed in LDS
    __syncthreads();          // all waves' tile-i writes visible; prev reads done
    if (i + 1 < steps) load_tile((i + 1) << 5, As[(i + 1) & 1], Bs[(i + 1) & 1]);
    const h16* Asb = As[i & 1];
    const h16* Bsb = Bs[i & 1];
    v16h bf0 = frag_b32x16(Bsb, LDB, wn * 32, lane);
    v16h bf1 = frag_b32x16(Bsb, LDB, wn * 32 + 16, lane);
#pragma unroll
    for (int mt = 0; mt < 4; ++mt) {
      v16h af = frag_a16x32(Asb, LDA, wm * 64 + mt * 16, lane);
      acc[mt][0] = wmma_f16(af, bf0, acc[mt][0]);
      acc[mt][1] = wmma_f16(af, bf1, acc[mt][1]);
    }
  }

  const int lrow = (lane >> 4) << 3;  // 0 or 8
  const int lcol = lane & 15;
#pragma unroll
  for (int mt = 0; mt < 4; ++mt)
#pragma unroll
    for (int nt = 0; nt < 2; ++nt) {
      int gn = bn + wn * 32 + nt * 16 + lcol;
      if (gn >= N) continue;
      float bv = bias ? bias[gn] : 0.f;
      int gm0 = bm + wm * 64 + mt * 16 + lrow;
#pragma unroll
      for (int r = 0; r < 8; ++r) {
        int gm = gm0 + r;
        float v = acc[mt][nt][r] + bv;
        if (res) v += res[(size_t)gm * N + gn];
        if (ACT == ACT_RELU) v = fmaxf(v, 0.f);
        if (outf) outf[(size_t)gm * N + gn] = v;
        if (outh) outh[(size_t)gm * N + gn] = (h16)v;
      }
    }
}

// ---------------- fused top-k attention: per (b,h,16-row chunk) ----------------
// scores = q@k^T (WMMA, K=64); exact 10th-largest threshold; softmax(s/8); out = P@v (WMMA, K=512)
__global__ __launch_bounds__(256)
void gattn_kernel(const h16* __restrict__ qg, const h16* __restrict__ kg,
                  const h16* __restrict__ vg, h16* __restrict__ outg)
{
  __shared__ h16  kl[512 * 64];     // k[s][e]   64 KB
  __shared__ h16  vl[512 * 64];     // v[s][e]   64 KB
  __shared__ float sc[16 * 512];    // scores    32 KB (re-used as f16 probs)

  const int tid = threadIdx.x;
  const int wid = tid >> 5, lane = tid & 31;
  const int chunk = blockIdx.x & 31;          // C/16 = 32 chunks
  const int h = (blockIdx.x >> 5) & 7;
  const int b = blockIdx.x >> 8;
  const int r0 = chunk * 16;
  const size_t baseRow = (size_t)b * Cc;
  const int hE = h * Ee;
  const int lrow = (lane >> 4) << 3;
  const int lcol = lane & 15;

#ifdef HAS_TDM
  if (tid < 32) {                   // wave 0 drives the Tensor Data Mover
    tdm_load_tile_512x64(kg + baseRow * Dd + hE, kl);
    tdm_load_tile_512x64(vg + baseRow * Dd + hE, vl);
    WAIT_TENSOR0();
  }
  __syncthreads();
#else
  for (int c = tid; c < 4096; c += 256) {
    int s = c >> 3, ec = (c & 7) << 3;
    *(float4*)(&kl[s * 64 + ec]) = *(const float4*)(kg + (baseRow + s) * Dd + hE + ec);
    *(float4*)(&vl[s * 64 + ec]) = *(const float4*)(vg + (baseRow + s) * Dd + hE + ec);
  }
  __syncthreads();
#endif

  // ---- scores 16 x 512 : 32 col-tiles, 4 per wave; A frags straight from global q ----
  const h16* qtile = qg + (baseRow + r0) * Dd + hE;
#pragma unroll
  for (int j = 0; j < 4; ++j) {
    int ct = wid + 8 * j;
    v8f acc = {};
#pragma unroll
    for (int ks = 0; ks < 64; ks += 32) {
      v16h a = frag_a16x32(qtile + ks, Dd, 0, lane);
      v16h bf = frag_b32x16(kl + ks, 64, ct * 16, lane);
      acc = wmma_f16(a, bf, acc);
    }
#pragma unroll
    for (int r = 0; r < 8; ++r)
      sc[(lrow + r) * 512 + ct * 16 + lcol] = acc[r];
  }
  __syncthreads();

  // ---- per-row top-k threshold + softmax (2 rows per wave), probs into registers ----
  float pvr[2][16];
  for (int rr = 0; rr < 2; ++rr) {
    int row = wid * 2 + rr;
    float loc[16];
#pragma unroll
    for (int j = 0; j < 16; ++j) loc[j] = sc[row * 512 + lane + 32 * j];
    float rowmax = 0.f, thr = 0.f;
    for (int it = 0; it < KTOP; ++it) {
      float lm = -INFINITY; int lj = -1;
#pragma unroll
      for (int j = 0; j < 16; ++j) if (loc[j] > lm) { lm = loc[j]; lj = j; }
      float wm = wave_rmax(lm);
      if (it == 0) rowmax = wm;
      thr = wm;
      unsigned long long mk = __ballot(lm == wm);
      int leader = __ffsll(mk) - 1;
      bool amleader = (lane == leader);
#pragma unroll
      for (int j = 0; j < 16; ++j)
        if (amleader && j == lj) loc[j] = -INFINITY;   // remove one instance
    }
    float sum = 0.f;
#pragma unroll
    for (int j = 0; j < 16; ++j) {
      float s = sc[row * 512 + lane + 32 * j];
      float p = (s >= thr) ? __expf((s - rowmax) * 0.125f) : 0.f;  // 1/sqrt(64)
      pvr[rr][j] = p; sum += p;
    }
    sum = wave_rsum(sum);
    float inv = 1.f / sum;
#pragma unroll
    for (int j = 0; j < 16; ++j) pvr[rr][j] *= inv;
  }
  __syncthreads();
  // write f16 probs in place over the score buffer
  h16* prv = reinterpret_cast<h16*>(sc);
  for (int rr = 0; rr < 2; ++rr) {
    int row = wid * 2 + rr;
#pragma unroll
    for (int j = 0; j < 16; ++j)
      prv[row * 512 + lane + 32 * j] = (h16)pvr[rr][j];
  }
  __syncthreads();

  // ---- out = P(16x512) @ v(512x64) : 4 tiles, waves 0..3 ----
  if (wid < 4) {
    v8f acc = {};
    for (int ks = 0; ks < 512; ks += 32) {
      v16h a = frag_a16x32(prv + ks, 512, 0, lane);
      v16h bf = frag_b32x16_kn(vl + (size_t)ks * 64, 64, wid * 16, lane);
      acc = wmma_f16(a, bf, acc);
    }
#pragma unroll
    for (int r = 0; r < 8; ++r) {
      int row = r0 + lrow + r;
      int e = wid * 16 + lcol;
      outg[(baseRow + row) * Dd + hE + e] = (h16)acc[r];
    }
  }
}

// ---------------- patchwise attention core (P=16), one wave per row ----------------
template<int LQ, int LS>
__global__ __launch_bounds__(256)
void pwattn_kernel(const h16* __restrict__ q, const h16* __restrict__ kb,
                   const h16* __restrict__ vb, h16* __restrict__ outV)
{
  __shared__ h16 kl[8][LS * 16];
  __shared__ h16 vl[8][LS * 16];
  __shared__ float pl[8][LS];
  const int tid = threadIdx.x;
  const int wid = tid >> 5, lane = tid & 31;
  const size_t n = (size_t)blockIdx.x * 8 + wid;
  const h16* ksrc = kb + n * (LS * 16);
  const h16* vsrc = vb + n * (LS * 16);
  for (int i = lane; i < LS * 2; i += 32) {
    *(float4*)(&kl[wid][i * 8]) = *(const float4*)(ksrc + i * 8);
    *(float4*)(&vl[wid][i * 8]) = *(const float4*)(vsrc + i * 8);
  }
  __syncthreads();
  const h16* qrow = q + n * (LQ * 16);
  const int e16 = lane & 15, half = lane >> 4;
  for (int lq = 0; lq < LQ; ++lq) {
    float qv = (float)qrow[lq * 16 + e16];      // lanes 0..15 hold q[lq][e]
    constexpr int CPL = LS / 32;
    float s[CPL];
#pragma unroll
    for (int c = 0; c < CPL; ++c) {
      int ls = lane + 32 * c;
      float a = 0.f;
#pragma unroll
      for (int e = 0; e < 16; ++e)
        a += __shfl(qv, e, 32) * (float)kl[wid][ls * 16 + e];
      s[c] = a * 0.25f;                         // 1/sqrt(P)
    }
    float m = s[0];
#pragma unroll
    for (int c = 1; c < CPL; ++c) m = fmaxf(m, s[c]);
    m = wave_rmax(m);
    float sum = 0.f, p[CPL];
#pragma unroll
    for (int c = 0; c < CPL; ++c) { p[c] = __expf(s[c] - m); sum += p[c]; }
    sum = wave_rsum(sum);
    float inv = 1.f / sum;
#pragma unroll
    for (int c = 0; c < CPL; ++c) pl[wid][lane + 32 * c] = p[c] * inv;
    __syncthreads();
    float va = 0.f;
    const int ls0 = half * (LS / 2);
    for (int ls = ls0; ls < ls0 + LS / 2; ++ls)
      va += pl[wid][ls] * (float)vl[wid][ls * 16 + e16];
    va += __shfl_xor(va, 16, 32);
    if (lane < 16) outV[n * (LQ * 16) + lq * 16 + lane] = (h16)va;
    __syncthreads();
  }
}

// ---------------- instance norm over time + transpose to (B*C, S) f16 ----------------
__global__ __launch_bounds__(256)
void instnorm_kernel(const float* __restrict__ xe, h16* __restrict__ xT, float* __restrict__ ms)
{
  const int tid = threadIdx.x;
  const int wid = tid >> 5, lane = tid & 31;
  const int bc = blockIdx.x;
  const int b = bc >> 9, c = bc & 511;
  const float* base = xe + (size_t)b * Ss * Cc + c;
  float x0 = base[(size_t)tid * Cc];
  float x1 = base[(size_t)(tid + 256) * Cc];
  float sum = x0 + x1, sq = x0 * x0 + x1 * x1;
  sum = wave_rsum(sum); sq = wave_rsum(sq);
  __shared__ float r1[8], r2[8], stat[2];
  if (lane == 0) { r1[wid] = sum; r2[wid] = sq; }
  __syncthreads();
  if (tid == 0) {
    float s = 0.f, qq = 0.f;
    for (int i = 0; i < 8; ++i) { s += r1[i]; qq += r2[i]; }
    float mean = s / 512.f;
    float var = qq / 512.f - mean * mean;
    float sd = sqrtf(var + 1e-5f);
    stat[0] = mean; stat[1] = 1.f / sd;
    ms[2 * bc] = mean; ms[2 * bc + 1] = sd;
  }
  __syncthreads();
  float mean = stat[0], isd = stat[1];
  xT[(size_t)bc * Ss + tid]       = (h16)((x0 - mean) * isd);
  xT[(size_t)bc * Ss + tid + 256] = (h16)((x1 - mean) * isd);
}

// ---------------- sinusoidal PE add (+f16 copy) ----------------
__global__ __launch_bounds__(256)
void pe_add_kernel(float* __restrict__ emb, h16* __restrict__ embh)
{
  size_t i = (size_t)blockIdx.x * 256 + threadIdx.x;  // over B*C*D
  int d = (int)(i & 511);
  int c = (int)((i >> 9) & 511);
  int d2 = d & ~1;
  float div = __expf((float)d2 * (-9.210340371976184f / 512.f)); // -ln(1e4)/D
  float ang = (float)c * div;
  float pe = (d & 1) ? __cosf(ang) : __sinf(ang);
  float v = emb[i] + pe;
  emb[i] = v;
  embh[i] = (h16)v;
}

// ---------------- leaky_relu(0.5) -> f16 ----------------
__global__ void leaky_to_h(const float* __restrict__ in, h16* __restrict__ out, size_t nel) {
  size_t i = (size_t)blockIdx.x * 256 + threadIdx.x;
  if (i < nel) { float v = in[i]; out[i] = (h16)(v > 0.f ? v : 0.5f * v); }
}

// ---------------- elu + layernorm(DF) -> f16 ----------------
__global__ __launch_bounds__(256)
void elu_ln_kernel(const float* __restrict__ hf, const float* __restrict__ g,
                   const float* __restrict__ bt, h16* __restrict__ out)
{
  const int tid = threadIdx.x;
  const int wid = tid >> 5, lane = tid & 31;
  const size_t n = blockIdx.x;
  const float* row = hf + n * (size_t)DFf;
  float loc[8];
  float sum = 0.f, sq = 0.f;
#pragma unroll
  for (int i = 0; i < 8; ++i) {
    float x = row[tid + 256 * i];
    float e = x > 0.f ? x : (__expf(x) - 1.f);
    loc[i] = e; sum += e; sq += e * e;
  }
  sum = wave_rsum(sum); sq = wave_rsum(sq);
  __shared__ float r1[8], r2[8], stat[2];
  if (lane == 0) { r1[wid] = sum; r2[wid] = sq; }
  __syncthreads();
  if (tid == 0) {
    float s = 0.f, qq = 0.f;
    for (int i = 0; i < 8; ++i) { s += r1[i]; qq += r2[i]; }
    float mu = s / (float)DFf;
    float var = qq / (float)DFf - mu * mu;
    stat[0] = mu; stat[1] = rsqrtf(var + 1e-5f);
  }
  __syncthreads();
  float mu = stat[0], rstd = stat[1];
#pragma unroll
  for (int i = 0; i < 8; ++i) {
    int d = tid + 256 * i;
    out[n * (size_t)DFf + d] = (h16)((loc[i] - mu) * rstd * g[d] + bt[d]);
  }
}

// ---------------- f32 -> f16 ----------------
__global__ void f32_to_f16_kernel(const float* __restrict__ in, h16* __restrict__ out, size_t nel) {
  size_t i = (size_t)blockIdx.x * 256 + threadIdx.x;
  if (i < nel) out[i] = (h16)in[i];
}

// ---------------- final transpose + denorm ----------------
__global__ void final_kernel(const float* __restrict__ dec, const float* __restrict__ ms,
                             float* __restrict__ out)
{
  size_t i = (size_t)blockIdx.x * 256 + threadIdx.x;  // over B*PRED*C
  int c = (int)(i & 511);
  int p = (int)((i >> 9) % PREDp);
  int b = (int)(i / ((size_t)PREDp * Cc));
  int bc = b * Cc + c;
  out[i] = dec[(size_t)bc * PREDp + p] * ms[2 * bc + 1] + ms[2 * bc];
}

// =============================== host ===============================
extern "C" void kernel_launch(void* const* d_in, const int* in_sizes, int n_in,
                              void* d_out, int out_size, void* d_ws, size_t ws_size,
                              hipStream_t stream)
{
  (void)in_sizes; (void)n_in; (void)out_size; (void)ws_size;
  const float* x_enc = (const float*)d_in[0];

  // ---- workspace carve ----
  char* wp = (char*)d_ws;
  auto alloc = [&](size_t bytes) -> void* {
    void* r = (void*)wp;
    wp += (bytes + 255) & ~(size_t)255;
    return r;
  };
  auto hbuf = [&](size_t nel) { return (h16*)alloc(nel * 2); };
  auto fbuf = [&](size_t nel) { return (float*)alloc(nel * 4); };

  h16* w_embed = hbuf(512 * 512);
  h16* w_q   = hbuf(512 * 512);
  h16* w_k   = hbuf(512 * 512);
  h16* w_v   = hbuf(512 * 512);
  h16* w_o   = hbuf(512 * 512);
  h16* w_enc = hbuf(512 * 512);
  h16* w_f1q = hbuf(512 * 2048);
  h16* w_f1k = hbuf(512 * 512);
  h16* w_f1v = hbuf(512 * 512);
  h16* w_f1o = hbuf((size_t)2048 * 2048);
  h16* w_f2q = hbuf((size_t)2048 * 512);
  h16* w_f2k = hbuf((size_t)2048 * 2048);
  h16* w_f2v = hbuf((size_t)2048 * 2048);
  h16* w_f2o = hbuf(512 * 512);
  h16* w_fc  = hbuf(512 * 96);

  h16*   xT    = hbuf((size_t)NR * 512);
  float* ms    = fbuf((size_t)NR * 2);
  float* embf  = fbuf((size_t)NR * 512);
  h16*   embh  = hbuf((size_t)NR * 512);
  h16*   qh    = hbuf((size_t)NR * 512);
  h16*   kh    = hbuf((size_t)NR * 512);
  h16*   vh    = hbuf((size_t)NR * 512);
  h16*   attnh = hbuf((size_t)NR * 512);
  float* yf    = fbuf((size_t)NR * 512);
  h16*   ylkh  = hbuf((size_t)NR * 512);
  h16*   x1h   = hbuf((size_t)NR * 512);
  float* q1f   = fbuf((size_t)NR * 2048);
  h16*   q1h   = hbuf((size_t)NR * 2048);
  h16*   k1h   = hbuf((size_t)NR * 512);
  h16*   v1h   = hbuf((size_t)NR * 512);
  h16*   V1h   = hbuf((size_t)NR * 2048);
  float* hf    = fbuf((size_t)NR * 2048);
  h16*   hnh   = hbuf((size_t)NR * 2048);
  float* q2f   = fbuf((size_t)NR * 512);
  h16*   q2h   = hbuf((size_t)NR * 512);
  h16*   k2h   = hbuf((size_t)NR * 2048);
  h16*   v2h   = hbuf((size_t)NR * 2048);
  h16*   V2h   = hbuf((size_t)NR * 512);
  h16*   yfh   = hbuf((size_t)NR * 512);
  float* decf  = fbuf((size_t)NR * 96);

  auto cv = [&](const void* src, h16* dst, size_t nel) {
    f32_to_f16_kernel<<<dim3((unsigned)((nel + 255) / 256)), dim3(256), 0, stream>>>(
        (const float*)src, dst, nel);
  };
  cv(d_in[4],  w_embed, 512 * 512);
  cv(d_in[5],  w_q,   512 * 512);
  cv(d_in[7],  w_k,   512 * 512);
  cv(d_in[9],  w_v,   512 * 512);
  cv(d_in[11], w_o,   512 * 512);
  cv(d_in[13], w_enc, 512 * 512);
  cv(d_in[15], w_f1q, 512 * 2048);
  cv(d_in[17], w_f1k, 512 * 512);
  cv(d_in[19], w_f1v, 512 * 512);
  cv(d_in[21], w_f1o, (size_t)2048 * 2048);
  cv(d_in[23], w_f2q, (size_t)2048 * 512);
  cv(d_in[25], w_f2k, (size_t)2048 * 2048);
  cv(d_in[27], w_f2v, (size_t)2048 * 2048);
  cv(d_in[29], w_f2o, 512 * 512);
  cv(d_in[33], w_fc,  512 * 96);

  auto gemm = [&](int act, const h16* Aa, const h16* Ww, const float* bias, const float* res,
                  float* of, h16* oh, int M, int N, int K) {
    dim3 grid((N + 127) / 128, M / 128), blk(256);
    if (act == ACT_RELU)
      gemm_wmma_kernel<ACT_RELU><<<grid, blk, 0, stream>>>(Aa, Ww, bias, res, of, oh, M, N, K);
    else
      gemm_wmma_kernel<ACT_NONE><<<grid, blk, 0, stream>>>(Aa, Ww, bias, res, of, oh, M, N, K);
  };

  // 1. instance norm + transpose
  instnorm_kernel<<<NR, 256, 0, stream>>>(x_enc, xT, ms);
  // 2. embed = relu(xT @ embed_w), then + PE
  gemm(ACT_RELU, xT, w_embed, nullptr, nullptr, embf, nullptr, NR, 512, 512);
  pe_add_kernel<<<(NR * 512) / 256, 256, 0, stream>>>(embf, embh);
  // 3. q/k/v
  gemm(ACT_RELU, embh, w_q, (const float*)d_in[6],  nullptr, nullptr, qh, NR, 512, 512);
  gemm(ACT_NONE, embh, w_k, (const float*)d_in[8],  nullptr, nullptr, kh, NR, 512, 512);
  gemm(ACT_RELU, embh, w_v, (const float*)d_in[10], nullptr, nullptr, vh, NR, 512, 512);
  // 4. top-k attention
  gattn_kernel<<<Bb * Hh * 32, 256, 0, stream>>>(qh, kh, vh, attnh);
  // 5. y = emb + attn @ o_w + o_b ; leaky ; enc
  gemm(ACT_NONE, attnh, w_o, (const float*)d_in[12], embf, yf, nullptr, NR, 512, 512);
  leaky_to_h<<<(NR * 512) / 256, 256, 0, stream>>>(yf, ylkh, (size_t)NR * 512);
  gemm(ACT_NONE, ylkh, w_enc, (const float*)d_in[14], nullptr, nullptr, x1h, NR, 512, 512);
  // 6. pwattn FFN 1
  gemm(ACT_RELU, x1h, w_f1q, (const float*)d_in[16], nullptr, q1f, q1h, NR, 2048, 512);
  gemm(ACT_NONE, x1h, w_f1k, (const float*)d_in[18], nullptr, nullptr, k1h, NR, 512, 512);
  gemm(ACT_RELU, x1h, w_f1v, (const float*)d_in[20], nullptr, nullptr, v1h, NR, 512, 512);
  pwattn_kernel<128, 32><<<NR / 8, 256, 0, stream>>>(q1h, k1h, v1h, V1h);
  gemm(ACT_NONE, V1h, w_f1o, (const float*)d_in[22], q1f, hf, nullptr, NR, 2048, 2048);
  // 7. elu + layernorm
  elu_ln_kernel<<<NR, 256, 0, stream>>>(hf, (const float*)d_in[31], (const float*)d_in[32], hnh);
  // 8. pwattn FFN 2
  gemm(ACT_RELU, hnh, w_f2q, (const float*)d_in[24], nullptr, q2f, q2h, NR, 512, 2048);
  gemm(ACT_NONE, hnh, w_f2k, (const float*)d_in[26], nullptr, nullptr, k2h, NR, 2048, 2048);
  gemm(ACT_RELU, hnh, w_f2v, (const float*)d_in[28], nullptr, nullptr, v2h, NR, 2048, 2048);
  pwattn_kernel<32, 128><<<NR / 8, 256, 0, stream>>>(q2h, k2h, v2h, V2h);
  gemm(ACT_NONE, V2h, w_f2o, (const float*)d_in[30], q2f, nullptr, yfh, NR, 512, 512);
  // 9. fc + transpose + denorm
  gemm(ACT_NONE, yfh, w_fc, (const float*)d_in[34], nullptr, decf, nullptr, NR, 96, 512);
  final_kernel<<<(Bb * PREDp * Cc) / 256, 256, 0, stream>>>(decf, ms, (float*)d_out);
}
